// GroupWiseLinearProjector_79001628442639
// MI455X (gfx1250) — matrix operand order repaired and checked by reading, not compiled, not run on hardware
//
#include <hip/hip_runtime.h>

// ---------- types ----------
typedef __attribute__((ext_vector_type(16))) __bf16 v16bf;
typedef __attribute__((ext_vector_type(8)))  float  v8f;
typedef __attribute__((ext_vector_type(4)))  int    i32x4;
typedef __attribute__((ext_vector_type(4)))  unsigned u32x4;

#define B_   16
#define CS_  512
#define CT_  512
#define H_   64
#define W_   64
#define M_   4096   // B * (H/4) * (W/4)
#define K_   512
#define N_   512
#define G_   16
#define LP   72     // LDS row pitch in bf16 elements (144 B, 16B-aligned rows)
#define TILE_USHORT (128 * LP)
#define TILE_BYTES  (TILE_USHORT * 2)

union FragU { v16bf v; i32x4 q[2]; };

static __device__ __forceinline__ unsigned short f2bf(float f) {
  unsigned u = __float_as_uint(f);
  u += 0x7fffu + ((u >> 16) & 1u);   // round-to-nearest-even
  return (unsigned short)(u >> 16);
}

// flat shared pointer -> 32-bit LDS byte address (low 32 bits of the flat address)
static __device__ __forceinline__ unsigned lds_addr(const void* p) {
  return (unsigned)(unsigned long long)p;
}

// global -> LDS async copy, 16 bytes per lane (ASYNCcnt path)
static __device__ __forceinline__ void async_copy_b128(const void* gptr, unsigned loff) {
#if __has_builtin(__builtin_amdgcn_global_load_async_to_lds_b128)
  typedef __attribute__((address_space(1))) i32x4 gvec_t;
  typedef __attribute__((address_space(3))) i32x4 lvec_t;
  __builtin_amdgcn_global_load_async_to_lds_b128(
      (gvec_t*)(unsigned long long)gptr,
      (lvec_t*)(unsigned long long)loff,
      0, 0);
#else
  unsigned long long ga = (unsigned long long)gptr;
  asm volatile("global_load_async_to_lds_b128 %0, %1, off"
               :: "v"(loff), "v"(ga) : "memory");
#endif
}

static __device__ __forceinline__ void wait_async0() {
#if __has_builtin(__builtin_amdgcn_s_wait_asynccnt)
  __builtin_amdgcn_s_wait_asynccnt(0);
#else
  asm volatile("s_wait_asynccnt 0" ::: "memory");
#endif
}

// ---------- pass 1a: convert Wg (f32 [g][o][c]) -> bf16, same layout ----------
__global__ void __launch_bounds__(256) wcvt_kernel(const float* __restrict__ Wg,
                                                   unsigned short* __restrict__ Wb) {
  int i = (blockIdx.x * 256 + threadIdx.x) * 4;   // 4,194,304 total elems
  float4 v = *(const float4*)(Wg + i);
  unsigned r0 = (unsigned)f2bf(v.x) | ((unsigned)f2bf(v.y) << 16);
  unsigned r1 = (unsigned)f2bf(v.z) | ((unsigned)f2bf(v.w) << 16);
  unsigned* p = (unsigned*)(Wb + i);
  p[0] = r0; p[1] = r1;
}

// ---------- pass 1b: gather x (f32 [b][c][h][w]) -> bf16 Ax[g][m][c] ----------
__global__ void __launch_bounds__(256) xgather_kernel(const float* __restrict__ x,
                                                      unsigned short* __restrict__ Ax) {
  __shared__ alignas(16) unsigned short lds[64 * 34];  // 64 w rows, pitch 34 (conflict-free)
  int t  = threadIdx.x;
  int cb = blockIdx.x;   // c block: cb*32
  int h  = blockIdx.y;
  int b  = blockIdx.z;
  int tx = t & 15, ty = t >> 4;          // tx: w4 group, ty: c pair
  int c0 = cb * 32 + 2 * ty;

  const float* px = x + (((size_t)(b * CS_ + c0) * H_ + h) * W_) + 4 * tx;
  float4 va = *(const float4*)px;              // channel c0,   w = 4tx..4tx+3
  float4 vb = *(const float4*)(px + H_ * W_);  // channel c0+1
  const float* fa = &va.x;
  const float* fb = &vb.x;
#pragma unroll
  for (int j = 0; j < 4; ++j) {
    unsigned pk = (unsigned)f2bf(fa[j]) | ((unsigned)f2bf(fb[j]) << 16);
    *(unsigned*)&lds[(4 * tx + j) * 34 + 2 * ty] = pk;  // row = w, col = c pair
  }
  __syncthreads();

  int w = t >> 2, part = t & 3;          // 4 threads per w row, 16 B each
  unsigned d0 = *(const unsigned*)&lds[w * 34 + part * 8 + 0];
  unsigned d1 = *(const unsigned*)&lds[w * 34 + part * 8 + 2];
  unsigned d2 = *(const unsigned*)&lds[w * 34 + part * 8 + 4];
  unsigned d3 = *(const unsigned*)&lds[w * 34 + part * 8 + 6];
  u32x4 d = {d0, d1, d2, d3};

  int q = w & 3, w4 = w >> 2;
  int g = (h & 3) * 4 + q;
  int m = b * 256 + (h >> 2) * 16 + w4;
  *(u32x4*)(Ax + ((size_t)(g * M_ + m) * K_ + cb * 32 + part * 8)) = d;
}

// ---------- pass 2: per-phase GEMM  D[o, m] = W_g x X_g^T  (bf16 WMMA) ----------
// Double-buffered LDS tiles filled by GLOBAL_LOAD_ASYNC_TO_LDS_B128 (ASYNCcnt),
// one barrier per k-block, WMMA overlapped with the prefetch of the next block.
__global__ void __launch_bounds__(256) gemm_kernel(const unsigned short* __restrict__ Wb,
                                                   const unsigned short* __restrict__ Ax,
                                                   float* __restrict__ out) {
  __shared__ alignas(16) unsigned short sA[2][TILE_USHORT];  // W tiles: 128 o rows x 64 k
  __shared__ alignas(16) unsigned short sB[2][TILE_USHORT];  // X tiles: 128 m rows x 64 k

  int t  = threadIdx.x;
  int g  = blockIdx.z;
  int mT = blockIdx.x * 128;
  int oT = blockIdx.y * 128;
  const unsigned short* gA = Wb + (size_t)(g * N_ + oT) * K_;
  const unsigned short* gB = Ax + (size_t)(g * M_ + mT) * K_;

  int lane  = t & 31;
  int wave  = t >> 5;
  int waveO = (wave & 1) * 64;    // 2 wave rows cover 128 o
  int waveM = (wave >> 1) * 32;   // 4 wave cols cover 128 m
  int lm = lane & 15;
  int hi = lane >> 4;             // lane half selects K sub-range per ISA layout

  unsigned baseA = lds_addr(&sA[0][0]);
  unsigned baseB = lds_addr(&sB[0][0]);

  v8f acc[4][2];
  v8f zero = {0.f, 0.f, 0.f, 0.f, 0.f, 0.f, 0.f, 0.f};
#pragma unroll
  for (int i = 0; i < 4; ++i)
#pragma unroll
    for (int j = 0; j < 2; ++j) acc[i][j] = zero;

  // per-thread chunk assignment: 1024 16B chunks per 128x64 tile, coalesced
  // prefetch k-block 0 into buffer 0
#pragma unroll
  for (int i = 0; i < 4; ++i) {
    int id = i * 256 + t;
    int row = id >> 3, ch = id & 7;
    size_t goff = (size_t)row * K_ + ch * 8;
    unsigned loff = (unsigned)(row * (LP * 2) + ch * 16);
    async_copy_b128(gA + goff, baseA + loff);
    async_copy_b128(gB + goff, baseB + loff);
  }

  for (int kb = 0; kb < K_ / 64; ++kb) {
    int cur = kb & 1;
    wait_async0();      // this wave's async fills for block kb are in LDS
    __syncthreads();    // all waves done filling kb AND done reading block kb-1

    if (kb < K_ / 64 - 1) {           // prefetch block kb+1 into the other buffer
      int nxt = cur ^ 1;
#pragma unroll
      for (int i = 0; i < 4; ++i) {
        int id = i * 256 + t;
        int row = id >> 3, ch = id & 7;
        size_t goff = (size_t)row * K_ + (kb + 1) * 64 + ch * 8;
        unsigned loff = (unsigned)(nxt * TILE_BYTES + row * (LP * 2) + ch * 16);
        async_copy_b128(gA + goff, baseA + loff);
        async_copy_b128(gB + goff, baseB + loff);
      }
    }

#pragma unroll
    for (int ks = 0; ks < 2; ++ks) {
      FragU a[4], bfr[2];
      // A frag (16x32): lane<16 holds K 0..7 & 16..23; lane>=16 holds K 8..15 & 24..31
#pragma unroll
      for (int i = 0; i < 4; ++i) {
        const unsigned short* p = &sA[cur][(waveO + i * 16 + lm) * LP + ks * 32 + hi * 8];
        a[i].q[0] = *(const i32x4*)p;
        a[i].q[1] = *(const i32x4*)(p + 16);
      }
      // B frag (32x16): lane<16 holds K 0..7 & 16..23 of its column; lane>=16 holds K 8..15 & 24..31
#pragma unroll
      for (int j = 0; j < 2; ++j) {
        const unsigned short* p = &sB[cur][(waveM + j * 16 + lm) * LP + ks * 32 + hi * 8];
        bfr[j].q[0] = *(const i32x4*)p;
        bfr[j].q[1] = *(const i32x4*)(p + 16);
      }
#pragma unroll
      for (int i = 0; i < 4; ++i)
#pragma unroll
        for (int j = 0; j < 2; ++j)
          acc[i][j] = __builtin_amdgcn_wmma_f32_16x16x32_bf16(
              false, a[i].v, false, bfr[j].v, (short)0, acc[i][j], false, false);
    }
  }

  // epilogue: D lane layout: col N (=m) = lm, rows M (=o) = vgpr + hi*8
  int r = g >> 2, q = g & 3;
#pragma unroll
  for (int j = 0; j < 2; ++j) {
    int m  = mT + waveM + j * 16 + lm;
    int bb = m >> 8, h4 = (m >> 4) & 15, w4 = m & 15;
    size_t rowbase = ((size_t)bb * CT_) * (H_ * W_) + (size_t)(4 * h4 + r) * W_ + 4 * w4 + q;
#pragma unroll
    for (int i = 0; i < 4; ++i) {
      int obase = oT + waveO + i * 16 + hi * 8;
#pragma unroll
      for (int v = 0; v < 8; ++v) {
        out[rowbase + (size_t)(obase + v) * (H_ * W_)] = acc[i][j][v];
      }
    }
  }
}

// ---------- launcher ----------
extern "C" void kernel_launch(void* const* d_in, const int* in_sizes, int n_in,
                              void* d_out, int out_size, void* d_ws, size_t ws_size,
                              hipStream_t stream) {
  const float* x  = (const float*)d_in[0];   // [16, 512, 64, 64]
  const float* Wg = (const float*)d_in[1];   // [16, 512, 512]
  float* out = (float*)d_out;                // [16, 512, 64, 64]

  unsigned short* Ax = (unsigned short*)d_ws;                       // 64 MB: [16][4096][512] bf16
  unsigned short* Wb = (unsigned short*)((char*)d_ws +
                        (size_t)G_ * M_ * K_ * sizeof(unsigned short)); // 8 MB: [16][512][512] bf16

  wcvt_kernel<<<dim3((G_ * N_ * K_) / (256 * 4)), 256, 0, stream>>>(Wg, Wb);
  xgather_kernel<<<dim3(CS_ / 32, H_, B_), 256, 0, stream>>>(x, Ax);
  gemm_kernel<<<dim3(M_ / 128, N_ / 128, G_), 256, 0, stream>>>(Wb, Ax, out);
}
